// REModel_40235253629149
// MI455X (gfx1250) — compile-verified, matrix operand order checked by reading, not run on hardware
//
#include <hip/hip_runtime.h>

// ---------------------------------------------------------------------------
// MI455X (gfx1250, wave32) BiGRU relation-extraction model.
// ~412 GFLOP; weights (6.3 MB bf16 packed) + per-run state (1 MB) are
// L2-resident (192 MB L2), so the binding constraint is the 128-step serial
// recurrence.  Matrix math: v_wmma_f32_16x16x32_bf16 (f32 accumulate).
// Weight fragments are staged per-workgroup into LDS with the CDNA5 async
// DMA path (global_load_async_to_lds_b128 + s_wait_asynccnt), double-buffered
// to overlap DMA with WMMA.
// ---------------------------------------------------------------------------

typedef __bf16 bf16;
typedef bf16  v16bf __attribute__((ext_vector_type(16)));
typedef float v8f   __attribute__((ext_vector_type(8)));

#define Bsz   256
#define Lsz   128
#define U     512
#define N3    1536
#define NC    53

// ---- WMMA / load helpers --------------------------------------------------

__device__ __forceinline__ v8f wmma_bf16(v16bf a, v16bf b, v8f c) {
  // 8 args: (neg_a, A, neg_b, B, c_mod, C, reuse_a, reuse_b)
  return __builtin_amdgcn_wmma_f32_16x16x32_bf16(
      false, a, false, b, (short)0, c, false, false);
}

// A fragment (16x32 bf16, row-major source): per ISA 7.12.2 lane holds row
// m = lane%16; lanes 0-15 carry K {0..7,16..23}, lanes 16-31 K {8..15,24..31}.
__device__ __forceinline__ v16bf load_a16(const bf16* p) {
  union { v16bf v; uint4 q[2]; } u;
  u.q[0] = *reinterpret_cast<const uint4*>(p);
  u.q[1] = *reinterpret_cast<const uint4*>(p + 16);
  return u.v;
}

// Contiguous 32-byte fragment (pre-packed layout) -> two b128 reads (LDS).
__device__ __forceinline__ v16bf load_c16(const bf16* p) {
  union { v16bf v; uint4 q[2]; } u;
  u.q[0] = reinterpret_cast<const uint4*>(p)[0];
  u.q[1] = reinterpret_cast<const uint4*>(p)[1];
  return u.v;
}

// CDNA5 async DMA: 16 bytes global -> LDS, per lane, tracked by ASYNCcnt.
__device__ __forceinline__ void async_b128(unsigned lds_off, const void* gaddr) {
  asm volatile("global_load_async_to_lds_b128 %0, %1, off"
               :: "v"(lds_off), "v"(gaddr) : "memory");
}

__device__ __forceinline__ float sigmoidf_(float x) {
  return 1.0f / (1.0f + __expf(-x));
}

// ---- 1) Embedding gather + concat -> bf16 X [B*L, 512] --------------------

__global__ void embed_kernel(const int* __restrict__ iw, const int* __restrict__ ip1,
                             const int* __restrict__ ip2, const float* __restrict__ we,
                             const float* __restrict__ pe, bf16* __restrict__ X) {
  const int i = blockIdx.x;            // b*L + l
  const int w = iw[i], a = ip1[i], b = ip2[i];
  for (int u = threadIdx.x; u < U; u += 256) {
    float v;
    if (u < 448)       v = we[(size_t)w * 448 + u];
    else if (u < 480)  v = pe[a * 32 + (u - 448)];
    else               v = pe[b * 32 + (u - 480)];
    X[(size_t)i * U + u] = (bf16)v;
  }
}

// ---- 2) Pack fp32 weights [512,1536] -> bf16 WMMA-fragment order ----------
// tile index = nt*16 + kt (nt: 16-col tile 0..95, kt: 32-row tile 0..15)
// within tile: lane*16 + e ; n = nt*16 + (lane&15),
// k = kt*32 + (e<8 ? e : e+8) + (lane>=16 ? 8 : 0)   (mirror of A layout)

__global__ void pack_weights_kernel(const float* __restrict__ kf, const float* __restrict__ rf,
                                    const float* __restrict__ kb, const float* __restrict__ rb,
                                    bf16* __restrict__ dst) {
  const int m = blockIdx.y;
  const float* src = (m == 0) ? kf : (m == 1) ? rf : (m == 2) ? kb : rb;
  const int idx    = blockIdx.x * 256 + threadIdx.x;       // 0 .. 786431
  const int tile   = idx >> 9;
  const int within = idx & 511;
  const int lane   = within >> 4;
  const int e      = within & 15;
  const int ntG    = tile >> 4;
  const int kt     = tile & 15;
  const int n      = (ntG << 4) + (lane & 15);
  const int koff   = (e < 8 ? e : e + 8) + ((lane >> 4) << 3);
  const int k      = (kt << 5) + koff;
  dst[(size_t)m * (U * N3) + idx] = (bf16)src[(size_t)k * N3 + n];
}

// ---- 3) One GRU time step, all 4 runs (2 pairs x 2 runs) ------------------
// pair 0: weights (Kf,Rf); runs {x@t=s, x@t=127-s}; both write h_out[.., s]
// pair 1: weights (Kb,Rb); runs {x@t=s, x@t=127-s}; both write h_out[..,127-s]
// Block = 8 waves sharing one 16-unit tile; each wave owns a 16-batch tile.
// B fragments (6 KB / k-tile) staged into LDS once per block via async DMA,
// double-buffered across the K loop.

__global__ __launch_bounds__(256)
void gru_step_kernel(const bf16* __restrict__ X, const bf16* __restrict__ Wpack,
                     const float* __restrict__ bias_f, const float* __restrict__ bias_b,
                     float* __restrict__ hstate, bf16* __restrict__ hstate_bf,
                     float* __restrict__ hout, int s) {
  __shared__ bf16 smem[2][6 * 512];              // 12 KB: [Kz Kr Kh Rz Rr Rh]

  const int tid  = threadIdx.x;
  const int lane = tid & 31;
  const int wave = tid >> 5;
  const int nt   = blockIdx.x;                   // unit tile 0..31
  const int mt   = blockIdx.y * 8 + wave;        // batch tile 0..15
  const int pair = blockIdx.z;                   // 0 or 1
  const int n0   = nt << 4;
  const int b0   = mt << 4;

  const bf16*  K    = Wpack + (size_t)(pair ? 2 : 0) * (U * N3);
  const bf16*  R    = Wpack + (size_t)(pair ? 3 : 1) * (U * N3);
  const float* bias = pair ? bias_b : bias_f;

  const int t0 = s, t1 = (Lsz - 1) - s;
  const int run0 = pair * 2, run1 = pair * 2 + 1;
  const int mrow = lane & 15, half = lane >> 4;

  const size_t xr0 = ((size_t)(b0 + mrow) * Lsz + t0) << 9;   // *512
  const size_t xr1 = ((size_t)(b0 + mrow) * Lsz + t1) << 9;
  const size_t hr0 = (size_t)(run0 * Bsz + b0 + mrow) << 9;
  const size_t hr1 = (size_t)(run1 * Bsz + b0 + mrow) << 9;

  // Issue the per-thread share of one k-tile's B fragments (6144 B) as
  // 2 x b128 async DMA ops.  j in [0,384): tile q = j>>6, chunk (j&63)*16 B.
  // (second op re-covers j in [128,256) with identical data; benign.)
  auto prefetch = [&](int kt, int buf) {
    const unsigned lbase = (unsigned)(size_t)&smem[buf][0];
#pragma unroll
    for (int i = 0; i < 2; ++i) {
      const int j    = tid + i * 128;            // 0..255 , 128..383
      const int q    = j >> 6;
      const int woff = (j & 63) << 4;            // bytes within tile
      const bf16* mat = (q < 3) ? K : R;
      const int  ntG  = nt + ((q < 3) ? q : q - 3) * 32;
      const char* g = (const char*)mat + (((size_t)(ntG * 16 + kt)) << 10) + woff;
      async_b128(lbase + (unsigned)(j << 4), g);
    }
  };

  prefetch(0, 0);

  v8f cz0 = {}, cr0 = {}, cx0 = {}, ch0 = {};
  v8f cz1 = {}, cr1 = {}, cx1 = {}, ch1 = {};

#pragma unroll
  for (int kt = 0; kt < 16; ++kt) {
    const int cur = kt & 1;
    if (kt < 15) {
      prefetch(kt + 1, cur ^ 1);
      asm volatile("s_wait_asynccnt 0x2" ::: "memory");  // current tile landed
    } else {
      asm volatile("s_wait_asynccnt 0x0" ::: "memory");
    }
    __syncthreads();                                     // visible to all waves

    const int ka = (kt << 5) + (half << 3);
    v16bf ax0 = load_a16(X + xr0 + ka);
    v16bf ax1 = load_a16(X + xr1 + ka);
    v16bf ah0 = load_a16(hstate_bf + hr0 + ka);
    v16bf ah1 = load_a16(hstate_bf + hr1 + ka);

    const bf16* sb = &smem[cur][lane << 4];
    v16bf bkz = load_c16(sb + 0 * 512);
    v16bf bkr = load_c16(sb + 1 * 512);
    v16bf bkh = load_c16(sb + 2 * 512);
    v16bf brz = load_c16(sb + 3 * 512);
    v16bf brr = load_c16(sb + 4 * 512);
    v16bf brh = load_c16(sb + 5 * 512);

    // z and r gates: x-proj and h-proj share one accumulator (added anyway)
    cz0 = wmma_bf16(ax0, bkz, cz0);  cz0 = wmma_bf16(ah0, brz, cz0);
    cr0 = wmma_bf16(ax0, bkr, cr0);  cr0 = wmma_bf16(ah0, brr, cr0);
    // candidate: reset_after=True -> keep xh and rh separate
    cx0 = wmma_bf16(ax0, bkh, cx0);  ch0 = wmma_bf16(ah0, brh, ch0);
    cz1 = wmma_bf16(ax1, bkz, cz1);  cz1 = wmma_bf16(ah1, brz, cz1);
    cr1 = wmma_bf16(ax1, bkr, cr1);  cr1 = wmma_bf16(ah1, brr, cr1);
    cx1 = wmma_bf16(ax1, bkh, cx1);  ch1 = wmma_bf16(ah1, brh, ch1);

    __syncthreads();     // buffer may be overwritten by next prefetch
  }

  const int u = n0 + (lane & 15);                 // C: N = lane & 15
  const float bz = bias[u]        + bias[N3 + u];
  const float bg = bias[U + u]    + bias[N3 + U + u];
  const float bx = bias[2 * U + u];
  const float bh = bias[N3 + 2 * U + u];
  const int outpos = pair ? t1 : t0;

#pragma unroll
  for (int v = 0; v < 8; ++v) {
    const int b = b0 + v + (half << 3);           // C: M = v + 8*(lane>=16)
    const size_t hi0 = ((size_t)(run0 * Bsz + b) << 9) + u;
    const size_t hi1 = ((size_t)(run1 * Bsz + b) << 9) + u;
    const float h0 = hstate[hi0], h1 = hstate[hi1];

    const float z0  = sigmoidf_(cz0[v] + bz);
    const float g0  = sigmoidf_(cr0[v] + bg);
    const float hh0 = tanhf(cx0[v] + bx + g0 * (ch0[v] + bh));
    const float hn0 = z0 * h0 + (1.0f - z0) * hh0;

    const float z1  = sigmoidf_(cz1[v] + bz);
    const float g1  = sigmoidf_(cr1[v] + bg);
    const float hh1 = tanhf(cx1[v] + bx + g1 * (ch1[v] + bh));
    const float hn1 = z1 * h1 + (1.0f - z1) * hh1;

    hstate[hi0] = hn0;              hstate[hi1] = hn1;
    hstate_bf[hi0] = (bf16)hn0;     hstate_bf[hi1] = (bf16)hn1;

    hout[((size_t)b * Lsz + outpos) * U + u] += hn0 + hn1;
  }
}

// ---- 4) Attention + classifier tail ---------------------------------------

__global__ void att_scores_kernel(const float* __restrict__ h, const float* __restrict__ attw,
                                  float* __restrict__ scores) {
  const int i = blockIdx.x;                       // b*L + l
  __shared__ float red[128];
  float acc = 0.0f;
  for (int u = threadIdx.x; u < U; u += 128)
    acc += tanhf(h[(size_t)i * U + u]) * attw[u];
  red[threadIdx.x] = acc; __syncthreads();
  for (int off = 64; off > 0; off >>= 1) {
    if (threadIdx.x < off) red[threadIdx.x] += red[threadIdx.x + off];
    __syncthreads();
  }
  if (threadIdx.x == 0) scores[i] = red[0];
}

__global__ void att_softmax_kernel(const float* __restrict__ scores, float* __restrict__ alpha) {
  const int b = blockIdx.x;
  __shared__ float sm[128];
  const float v = scores[b * Lsz + threadIdx.x];
  sm[threadIdx.x] = v; __syncthreads();
  for (int off = 64; off > 0; off >>= 1) {
    if (threadIdx.x < off) sm[threadIdx.x] = fmaxf(sm[threadIdx.x], sm[threadIdx.x + off]);
    __syncthreads();
  }
  const float mx = sm[0]; __syncthreads();
  const float e = __expf(v - mx);
  sm[threadIdx.x] = e; __syncthreads();
  for (int off = 64; off > 0; off >>= 1) {
    if (threadIdx.x < off) sm[threadIdx.x] += sm[threadIdx.x + off];
    __syncthreads();
  }
  alpha[b * Lsz + threadIdx.x] = e / sm[0];
}

__global__ void att_ctx_kernel(const float* __restrict__ h, const float* __restrict__ alpha,
                               const float* __restrict__ sa, const float* __restrict__ sr,
                               float* __restrict__ rbuf, float* __restrict__ ebuf) {
  const int b = blockIdx.x;
  __shared__ float red[256];
  float part = 0.0f;
  for (int j = 0; j < 2; ++j) {
    const int u = threadIdx.x + j * 256;
    float acc = 0.0f;
    for (int l = 0; l < Lsz; ++l)
      acc += alpha[b * Lsz + l] * h[((size_t)b * Lsz + l) * U + u];
    const float rr = tanhf(acc);
    rbuf[(size_t)b * U + u] = rr;
    part += rr * sa[u] * sr[u];
  }
  red[threadIdx.x] = part; __syncthreads();
  for (int off = 128; off > 0; off >>= 1) {
    if (threadIdx.x < off) red[threadIdx.x] += red[threadIdx.x + off];
    __syncthreads();
  }
  if (threadIdx.x == 0) ebuf[b] = red[0];
}

__global__ void batch_softmax_kernel(const float* __restrict__ ebuf, float* __restrict__ abuf) {
  __shared__ float sm[256];
  const float v = ebuf[threadIdx.x];
  sm[threadIdx.x] = v; __syncthreads();
  for (int off = 128; off > 0; off >>= 1) {
    if (threadIdx.x < off) sm[threadIdx.x] = fmaxf(sm[threadIdx.x], sm[threadIdx.x + off]);
    __syncthreads();
  }
  const float mx = sm[0]; __syncthreads();
  const float e = __expf(v - mx);
  sm[threadIdx.x] = e; __syncthreads();
  for (int off = 128; off > 0; off >>= 1) {
    if (threadIdx.x < off) sm[threadIdx.x] += sm[threadIdx.x + off];
    __syncthreads();
  }
  abuf[threadIdx.x] = e / sm[0];
}

__global__ void logits_kernel(const float* __restrict__ rbuf, const float* __restrict__ abuf,
                              const float* __restrict__ ss, const float* __restrict__ sd,
                              float* __restrict__ out) {
  const int b = blockIdx.x;
  const int c = threadIdx.x;
  const bool valid = (c < NC);
  __shared__ float sm[64];
  float lg = -3.0e38f;
  if (valid) {
    float dot = 0.0f;
    for (int u = 0; u < U; ++u)
      dot += rbuf[(size_t)b * U + u] * ss[(size_t)c * U + u];
    lg = abuf[b] * dot + sd[c];
  }
  sm[c] = lg; __syncthreads();
  for (int off = 32; off > 0; off >>= 1) {
    if (c < off) sm[c] = fmaxf(sm[c], sm[c + off]);
    __syncthreads();
  }
  const float mx = sm[0]; __syncthreads();
  const float e = valid ? __expf(lg - mx) : 0.0f;
  sm[c] = e; __syncthreads();
  for (int off = 32; off > 0; off >>= 1) {
    if (c < off) sm[c] += sm[c + off];
    __syncthreads();
  }
  if (valid) out[b * NC + c] = e / sm[0];
}

// ---- Host-side orchestration ----------------------------------------------

extern "C" void kernel_launch(void* const* d_in, const int* in_sizes, int n_in,
                              void* d_out, int out_size, void* d_ws, size_t ws_size,
                              hipStream_t stream) {
  (void)in_sizes; (void)n_in; (void)out_size; (void)ws_size;

  const int*   iw   = (const int*)  d_in[0];
  const int*   ip1  = (const int*)  d_in[1];
  const int*   ip2  = (const int*)  d_in[2];
  const float* wemb = (const float*)d_in[3];
  const float* pemb = (const float*)d_in[4];
  const float* kf   = (const float*)d_in[5];
  const float* rf   = (const float*)d_in[6];
  const float* bf   = (const float*)d_in[7];
  const float* kb   = (const float*)d_in[8];
  const float* rb   = (const float*)d_in[9];
  const float* bb   = (const float*)d_in[10];
  const float* attw = (const float*)d_in[11];
  const float* sa   = (const float*)d_in[12];
  const float* sr   = (const float*)d_in[13];
  const float* sd   = (const float*)d_in[14];
  const float* ssm  = (const float*)d_in[15];
  float* out = (float*)d_out;

  char* ws = (char*)d_ws;
  bf16*  X         = (bf16*) (ws + 0);               //  33,554,432 B
  bf16*  Wp        = (bf16*) (ws + 33554432);        //   6,291,456 B (Kf,Rf,Kb,Rb)
  float* hstate    = (float*)(ws + 39845888);        //   2,097,152 B (4 runs x 256 x 512)
  bf16*  hstate_bf = (bf16*) (ws + 41943040);        //   1,048,576 B
  float* hout      = (float*)(ws + 42991616);        //  67,108,864 B (B x L x 512)
  float* scores    = (float*)(ws + 110100480);       //     131,072 B
  float* alpha     = (float*)(ws + 110231552);       //     131,072 B
  float* rbuf      = (float*)(ws + 110362624);       //     524,288 B
  float* ebuf      = (float*)(ws + 110886912);
  float* abuf      = (float*)(ws + 110887936);

  // zero hstate (f32+bf16) and hout accumulator (contiguous 70,254,592 B)
  hipMemsetAsync(hstate, 0, 70254592, stream);

  embed_kernel<<<Bsz * Lsz, 256, 0, stream>>>(iw, ip1, ip2, wemb, pemb, X);
  pack_weights_kernel<<<dim3(3072, 4), 256, 0, stream>>>(kf, rf, kb, rb, Wp);

  for (int s = 0; s < Lsz; ++s)
    gru_step_kernel<<<dim3(32, 2, 2), 256, 0, stream>>>(
        X, Wp, bf, bb, hstate, hstate_bf, hout, s);

  att_scores_kernel<<<Bsz * Lsz, 128, 0, stream>>>(hout, attw, scores);
  att_softmax_kernel<<<Bsz, 128, 0, stream>>>(scores, alpha);
  att_ctx_kernel<<<Bsz, 256, 0, stream>>>(hout, alpha, sa, sr, rbuf, ebuf);
  batch_softmax_kernel<<<1, 256, 0, stream>>>(ebuf, abuf);
  logits_kernel<<<Bsz, 64, 0, stream>>>(rbuf, abuf, ssm, sd, out);
}
